// SmallMLP_4672924418230
// MI455X (gfx1250) — matrix operand order repaired and checked by reading, not compiled, not run on hardware
//
#include <hip/hip_runtime.h>
#include <hip/hip_bf16.h>
#include <math.h>

// ---------------------------------------------------------------------------
// CDNA5 (gfx1250) WMMA / TDM types
// ---------------------------------------------------------------------------
typedef __attribute__((ext_vector_type(16))) _Float16     v16h;
typedef __attribute__((ext_vector_type(4)))  _Float16     v4h;
typedef __attribute__((ext_vector_type(8)))  float        v8f;
typedef __attribute__((ext_vector_type(4)))  unsigned int u32x4;
typedef __attribute__((ext_vector_type(8)))  unsigned int u32x8;

#define B_ROWS   8192
#define D_IN     784
#define D_H      4096
#define D_OUT    10
#define TILE_M   128
#define TILE_N   128
#define TILE_K   64          // staged K per TDM tile (2 WMMA sub-steps)
#define K_TILES  13          // ceil(784/64); tail zero-filled by TDM OOB rule
#define LDS_ROW  72          // 64 f16 data + 8 f16 TDM pad (4 dwords) per row

// ---------------------------------------------------------------------------
// TDM: 2D tile load (TILE_M rows x TILE_K f16) global -> LDS, per D# spec:
//   group0: count=1 | lds_addr | global_addr[56:0] | type=2
//   group1: data_size=2B, pad_enable (interval 32 dwords, amount 4 dwords),
//           tensor dims = remaining-from-tile-origin (OOB reads return zero
//           => free zero-fill of the K tail), tile = TILE_K x TILE_M.
// Issued once per block by wave 0; tracked by TENSORcnt.
// ---------------------------------------------------------------------------
__device__ __forceinline__ void tdm_load_tile_f16(const _Float16* gbase,
                                                  unsigned int lds_addr,
                                                  unsigned int cols_remaining,
                                                  unsigned int rows_remaining,
                                                  unsigned int row_stride_elems)
{
    const unsigned long long ga = (unsigned long long)(uintptr_t)gbase;
    u32x4 g0;
    g0[0] = 1u;                                         // count=1 (valid user D#)
    g0[1] = lds_addr;                                   // LDS byte address
    g0[2] = (unsigned int)(ga & 0xffffffffull);         // global_addr[31:0]
    g0[3] = (unsigned int)((ga >> 32) & 0x01ffffffull)  // global_addr[56:32]
          | 0x80000000u;                                // type=2 ("image")
    u32x8 g1;
    g1[0] = (1u << 16)                                  // data_size: 2 bytes
          | (1u << 20)                                  // pad_enable
          | (4u << 22)                                  // pad_interval: 32 dwords
          | (3u << 25);                                 // pad_amount: 4 dwords
    g1[1] = (cols_remaining & 0xffffu) << 16;                        // tensor_dim0 lo
    g1[2] = (cols_remaining >> 16) | ((rows_remaining & 0xffffu) << 16);
    g1[3] = (rows_remaining >> 16) | ((unsigned int)TILE_K << 16);   // tile_dim0
    g1[4] = (unsigned int)TILE_M;                       // tile_dim1 (tile_dim2=0)
    g1[5] = row_stride_elems;                           // tensor_dim0_stride lo
    g1[6] = 0u;
    g1[7] = 0u;
    asm volatile("tensor_load_to_lds %0, %1" :: "s"(g0), "s"(g1) : "memory");
}

// ---------------------------------------------------------------------------
// Kernel 0: zero the global-max slot
// ---------------------------------------------------------------------------
__global__ void init_max_kernel(unsigned int* hmax_bits) {
    if (threadIdx.x == 0) *hmax_bits = 0u;
}

// ---------------------------------------------------------------------------
// Prepass A: x (f32) -> f16, 4 elements/thread (float4 in, 64-bit out)
// ---------------------------------------------------------------------------
__global__ __launch_bounds__(256)
void cvt_x_kernel(const float* __restrict__ x, _Float16* __restrict__ xh, int n4) {
    const int i = blockIdx.x * 256 + threadIdx.x;
    if (i < n4) {
        const float4 v = *reinterpret_cast<const float4*>(x + (size_t)i * 4);
        v4h o;
        o[0] = (_Float16)v.x; o[1] = (_Float16)v.y;
        o[2] = (_Float16)v.z; o[3] = (_Float16)v.w;
        *reinterpret_cast<v4h*>(xh + (size_t)i * 4) = o;
    }
}

// ---------------------------------------------------------------------------
// Prepass B: w1 (f32) -> ternary({-1,0,1}) f16 (exact in f16)
// ---------------------------------------------------------------------------
__global__ __launch_bounds__(256)
void tern_w1_kernel(const float* __restrict__ w, _Float16* __restrict__ wh, int n4) {
    const int i = blockIdx.x * 256 + threadIdx.x;
    if (i < n4) {
        const float4 v = *reinterpret_cast<const float4*>(w + (size_t)i * 4);
        v4h o;
        o[0] = (v.x > 0.1f) ? (_Float16)1.0f : ((v.x < -0.1f) ? (_Float16)-1.0f : (_Float16)0.0f);
        o[1] = (v.y > 0.1f) ? (_Float16)1.0f : ((v.y < -0.1f) ? (_Float16)-1.0f : (_Float16)0.0f);
        o[2] = (v.z > 0.1f) ? (_Float16)1.0f : ((v.z < -0.1f) ? (_Float16)-1.0f : (_Float16)0.0f);
        o[3] = (v.w > 0.1f) ? (_Float16)1.0f : ((v.w < -0.1f) ? (_Float16)-1.0f : (_Float16)0.0f);
        *reinterpret_cast<v4h*>(wh + (size_t)i * 4) = o;
    }
}

// ---------------------------------------------------------------------------
// Kernel 1: h = relu(xh @ w1h^T + b1) + global max|h|.
// 256 threads = 8 wave32 waves; block tile 128x128, wave tile 32x64.
// TDM double-buffered f16 LDS stage-in; fragment build is pure ds_load_b128
// (no conversion VALU in the hot loop); 16 WMMAs per stage.
// ---------------------------------------------------------------------------
__global__ __launch_bounds__(256)
void gemm1_wmma_kernel(const _Float16* __restrict__ xh,
                       const _Float16* __restrict__ w1h,
                       const float* __restrict__ b1,
                       float* __restrict__ h,
                       unsigned int* __restrict__ hmax_bits)
{
    __shared__ _Float16 Ah[2][TILE_M][LDS_ROW];   // f16 x tiles (TDM-filled)
    __shared__ _Float16 Bh[2][TILE_N][LDS_ROW];   // ternary f16 w1 tiles

    const int tid    = threadIdx.x;
    const int lane   = tid & 31;
    const int wid    = tid >> 5;       // 0..7
    const int wave_m = wid & 3;        // 4 waves along M, 32 rows each
    const int wave_n = wid >> 2;       // 2 waves along N, 64 cols each

    const int m0 = blockIdx.x * TILE_M;
    const int n0 = blockIdx.y * TILE_N;

    // low 32 bits of a flat LDS pointer are the LDS byte offset (ISA 10.2)
    const unsigned int ldsA0 = (unsigned int)(uintptr_t)&Ah[0][0][0];
    const unsigned int ldsA1 = (unsigned int)(uintptr_t)&Ah[1][0][0];
    const unsigned int ldsB0 = (unsigned int)(uintptr_t)&Bh[0][0][0];
    const unsigned int ldsB1 = (unsigned int)(uintptr_t)&Bh[1][0][0];

    v8f c[2][4];
    #pragma unroll
    for (int mf = 0; mf < 2; ++mf)
        #pragma unroll
        for (int nf = 0; nf < 4; ++nf)
            c[mf][nf] = v8f{};

    if (wid == 0) {
        tdm_load_tile_f16(xh  + (size_t)m0 * D_IN, ldsA0, D_IN, B_ROWS - m0, D_IN);
        tdm_load_tile_f16(w1h + (size_t)n0 * D_IN, ldsB0, D_IN, D_H    - n0, D_IN);
    }

    for (int kt = 0; kt < K_TILES; ++kt) {
        const int buf = kt & 1;
        if (wid == 0)
            __builtin_amdgcn_s_wait_tensorcnt(0);   // current buffer landed
        __syncthreads();                            // visible to all; everyone
                                                    // done reading buf^1
        if (wid == 0 && (kt + 1) < K_TILES) {
            const int k1 = (kt + 1) * TILE_K;
            tdm_load_tile_f16(xh  + (size_t)m0 * D_IN + k1, buf ? ldsA0 : ldsA1,
                              (unsigned)(D_IN - k1), (unsigned)(B_ROWS - m0), D_IN);
            tdm_load_tile_f16(w1h + (size_t)n0 * D_IN + k1, buf ? ldsB0 : ldsB1,
                              (unsigned)(D_IN - k1), (unsigned)(D_H - n0), D_IN);
        }

        // two 32-wide K sub-steps per staged 64-wide tile
        #pragma unroll
        for (int s = 0; s < 2; ++s) {
            const int kb = s * 32;

            // A fragments (16x32 f16 per ISA layout):
            // lanes 0-15: M=lane, halves 0..7 = K 0..7, 8..15 = K 16..23
            // lanes 16-31: M=lane-16, halves 0..7 = K 8..15, 8..15 = K 24..31
            const int ar  = lane & 15;
            const int akb = kb + ((lane < 16) ? 0 : 8);
            v16h a[2];
            #pragma unroll
            for (int mf = 0; mf < 2; ++mf) {
                const int row = wave_m * 32 + mf * 16 + ar;
                #pragma unroll
                for (int i = 0; i < 8; ++i) {
                    a[mf][i]     = Ah[buf][row][akb + i];
                    a[mf][8 + i] = Ah[buf][row][akb + 16 + i];
                }
            }

            // B fragments (32x16: N=lane&15, K split at lane 16)
            const int bn  = lane & 15;
            const int bkb = kb + ((lane < 16) ? 0 : 8);
            v16h b[4];
            #pragma unroll
            for (int nf = 0; nf < 4; ++nf) {
                const int coln = wave_n * 64 + nf * 16 + bn;
                #pragma unroll
                for (int i = 0; i < 8; ++i) {
                    b[nf][i]     = Bh[buf][coln][bkb + i];
                    b[nf][8 + i] = Bh[buf][coln][bkb + 16 + i];
                }
            }

            #pragma unroll
            for (int mf = 0; mf < 2; ++mf)
                #pragma unroll
                for (int nf = 0; nf < 4; ++nf)
                    c[mf][nf] = __builtin_amdgcn_wmma_f32_16x16x32_f16(
                        false, a[mf], false, b[nf], (short)0, c[mf][nf], false, false);
        }
    }

    // ---- epilogue: +bias, relu, store, track max ----
    // C/D layout: lanes 0-15 N=lane M=vgpr; lanes 16-31 N=lane-16 M=vgpr+8
    const int nl = lane & 15;
    const int mh = (lane < 16) ? 0 : 8;
    float lmax = 0.0f;
    #pragma unroll
    for (int mf = 0; mf < 2; ++mf) {
        #pragma unroll
        for (int nf = 0; nf < 4; ++nf) {
            const int nglob = n0 + wave_n * 64 + nf * 16 + nl;
            const float bias = b1[nglob];
            #pragma unroll
            for (int g = 0; g < 8; ++g) {
                const int mglob = m0 + wave_m * 32 + mf * 16 + mh + g;
                float v = c[mf][nf][g] + bias;
                v = fmaxf(v, 0.0f);
                h[(size_t)mglob * D_H + nglob] = v;
                lmax = fmaxf(lmax, v);
            }
        }
    }
    #pragma unroll
    for (int off = 16; off > 0; off >>= 1)
        lmax = fmaxf(lmax, __shfl_xor(lmax, off, 32));
    if (lane == 0)
        atomicMax(hmax_bits, __float_as_uint(lmax));   // h >= 0: bit order == float order
}

// ---------------------------------------------------------------------------
// Kernel 2: fake-quantize h, logits = hq @ ternary(w2)^T + b2, log_softmax.
// One wave per row (grid-strided); ternary w2 in LDS as int8 (40 KB).
// float4 / char4 vectorized (memory-bound pass; h likely resident in 192MB L2).
// ---------------------------------------------------------------------------
__global__ __launch_bounds__(256)
void head_kernel(const float* __restrict__ h,
                 const float* __restrict__ w2,
                 const float* __restrict__ b2,
                 const unsigned int* __restrict__ hmax_bits,
                 float* __restrict__ out)
{
    __shared__ signed char w2s[D_OUT * D_H];   // 40960 bytes
    for (int i = threadIdx.x; i < D_OUT * D_H; i += 256) {
        const float wv = w2[i];
        w2s[i] = (wv > 0.1f) ? (signed char)1
                             : ((wv < -0.1f) ? (signed char)-1 : (signed char)0);
    }
    __syncthreads();

    const float mx        = __uint_as_float(*hmax_bits);
    const float scale     = (mx > 0.0f) ? (127.0f / mx) : 1.0f;
    const float inv_scale = 1.0f / scale;

    const int lane = threadIdx.x & 31;
    const int wid  = threadIdx.x >> 5;

    for (int row = blockIdx.x * 8 + wid; row < B_ROWS; row += gridDim.x * 8) {
        float acc[D_OUT];
        #pragma unroll
        for (int o = 0; o < D_OUT; ++o) acc[o] = 0.0f;

        const float* hrow = h + (size_t)row * D_H;
        for (int ci = lane * 4; ci < D_H; ci += 128) {
            const float4 hv = *reinterpret_cast<const float4*>(hrow + ci);
            float hq[4];
            {
                float q0 = fminf(fmaxf(rintf(hv.x * scale), -128.0f), 127.0f);
                float q1 = fminf(fmaxf(rintf(hv.y * scale), -128.0f), 127.0f);
                float q2 = fminf(fmaxf(rintf(hv.z * scale), -128.0f), 127.0f);
                float q3 = fminf(fmaxf(rintf(hv.w * scale), -128.0f), 127.0f);
                hq[0] = q0 * inv_scale; hq[1] = q1 * inv_scale;
                hq[2] = q2 * inv_scale; hq[3] = q3 * inv_scale;
            }
            #pragma unroll
            for (int o = 0; o < D_OUT; ++o) {
                const char4 w4 = *reinterpret_cast<const char4*>(&w2s[o * D_H + ci]);
                acc[o] += hq[0] * (float)w4.x + hq[1] * (float)w4.y
                        + hq[2] * (float)w4.z + hq[3] * (float)w4.w;
            }
        }

        #pragma unroll
        for (int o = 0; o < D_OUT; ++o)
            #pragma unroll
            for (int off = 16; off > 0; off >>= 1)
                acc[o] += __shfl_xor(acc[o], off, 32);

        if (lane == 0) {
            float lg[D_OUT];
            float m = -INFINITY;
            #pragma unroll
            for (int o = 0; o < D_OUT; ++o) {
                lg[o] = acc[o] + b2[o];
                m = fmaxf(m, lg[o]);
            }
            float s = 0.0f;
            #pragma unroll
            for (int o = 0; o < D_OUT; ++o) s += __expf(lg[o] - m);
            const float lse = logf(s);
            #pragma unroll
            for (int o = 0; o < D_OUT; ++o)
                out[(size_t)row * D_OUT + o] = lg[o] - m - lse;
        }
    }
}

// ---------------------------------------------------------------------------
// launch
// ---------------------------------------------------------------------------
extern "C" void kernel_launch(void* const* d_in, const int* in_sizes, int n_in,
                              void* d_out, int out_size, void* d_ws, size_t ws_size,
                              hipStream_t stream) {
    const float* x  = (const float*)d_in[0];   // [8192, 784]
    const float* w1 = (const float*)d_in[1];   // [4096, 784]
    const float* b1 = (const float*)d_in[2];   // [4096]
    const float* w2 = (const float*)d_in[3];   // [10, 4096]
    const float* b2 = (const float*)d_in[4];   // [10]
    float* out = (float*)d_out;                // [8192, 10]

    // workspace layout (256B-aligned sections):
    //   [0]        : uint max bits
    //   [256]      : xh  f16 [8192,784]  (12,845,056 B)
    //   [+12.25MB] : w1h f16 [4096,784]  ( 6,422,528 B)
    //   [+ 6.12MB] : h   f32 [8192,4096] (134,217,728 B)
    char* ws = (char*)d_ws;
    unsigned int* hmax = (unsigned int*)ws;
    _Float16* xh  = (_Float16*)(ws + 256);
    _Float16* w1h = (_Float16*)(ws + 256 + (size_t)B_ROWS * D_IN * 2);
    float*    h   = (float*)(ws + 256 + (size_t)B_ROWS * D_IN * 2
                                      + (size_t)D_H * D_IN * 2);

    init_max_kernel<<<1, 32, 0, stream>>>(hmax);

    const int nx4 = (B_ROWS * D_IN) / 4;       // 1,605,632
    const int nw4 = (D_H * D_IN) / 4;          //   802,816
    cvt_x_kernel  <<<(nx4 + 255) / 256, 256, 0, stream>>>(x,  xh,  nx4);
    tern_w1_kernel<<<(nw4 + 255) / 256, 256, 0, stream>>>(w1, w1h, nw4);

    dim3 g1(B_ROWS / TILE_M, D_H / TILE_N);    // 64 x 32 blocks
    gemm1_wmma_kernel<<<g1, 256, 0, stream>>>(xh, w1h, b1, h, hmax);

    head_kernel<<<128, 256, 0, stream>>>(h, w2, b2, hmax, out);
}
